// TreeConvolutionLayer_36326833389616
// MI455X (gfx1250) — matrix operand order repaired and checked by reading, not compiled, not run on hardware
//
#include <hip/hip_runtime.h>

typedef _Float16 f16;
typedef __attribute__((ext_vector_type(4)))  _Float16 v4h;
typedef __attribute__((ext_vector_type(8)))  _Float16 v8h;
typedef __attribute__((ext_vector_type(16))) _Float16 v16h;
typedef __attribute__((ext_vector_type(4)))  float v4f;
typedef __attribute__((ext_vector_type(8)))  float v8f;

#define WAVES_PER_BLOCK 4
#define THREADS (WAVES_PER_BLOCK * 32)
#define WPAD 200   // padded K-stride (halves) for concat weights [64 filters][192 K]
#define PPAD 72    // padded F-stride (halves) for proj weights   [64 out][64 F]
#define HPAD 72    // padded stride for per-wave h staging        [32 rows][64 F]

static __device__ __forceinline__ v16h cat4(v4h a0, v4h a1, v4h a2, v4h a3) {
  v8h lo = __builtin_shufflevector(a0, a1, 0,1,2,3,4,5,6,7);
  v8h hi = __builtin_shufflevector(a2, a3, 0,1,2,3,4,5,6,7);
  return __builtin_shufflevector(lo, hi, 0,1,2,3,4,5,6,7,8,9,10,11,12,13,14,15);
}

// B fragment: lane holds column n; halves 0..7 = K p..p+7, halves 8..15 = K p+16..p+23.
static __device__ __forceinline__ v16h load_b_frag(const f16* p) {
  const v4h* q = (const v4h*)p;
  return cat4(q[0], q[1], q[4], q[5]);
}

// A fragment for one 32-wide K chunk of a row: halves 0..7 = K colbase..+7,
// halves 8..15 = K colbase+16..+23 (lane-half offset already folded into colbase).
static __device__ __forceinline__ v16h build_a_frag(const float* row, int colbase) {
  v4f p0 = *(const v4f*)(row + colbase);
  v4f p1 = *(const v4f*)(row + colbase + 4);
  v4f p2 = *(const v4f*)(row + colbase + 16);
  v4f p3 = *(const v4f*)(row + colbase + 20);
  return cat4(__builtin_convertvector(p0, v4h), __builtin_convertvector(p1, v4h),
              __builtin_convertvector(p2, v4h), __builtin_convertvector(p3, v4h));
}

// Each wave processes a PAIR of 16-node tiles (M=32) per iteration so every
// B fragment fetched from LDS feeds two WMMAs.
__global__ void __launch_bounds__(THREADS)
treeconv_wmma(const float* __restrict__ nodes,
              const long long* __restrict__ lidx,
              const long long* __restrict__ ridx,
              const float* __restrict__ Wp,
              const float* __restrict__ Wl,
              const float* __restrict__ Wr,
              const float* __restrict__ bias,
              const float* __restrict__ Wproj,
              const float* __restrict__ bproj,
              float* __restrict__ out,
              int numPairs)
{
  __shared__ f16   sW[64 * WPAD];              // Wcat as [f][k], k = 0..191
  __shared__ f16   sP[64 * PPAD];              // Wproj as [o][f]
  __shared__ float sBias[64];
  __shared__ float sBproj[64];
  __shared__ f16   sH[WAVES_PER_BLOCK][32 * HPAD];

  const int tid = threadIdx.x;

  // ---- stage weights to LDS (once per block), f32 -> f16 ----
  for (int i = tid; i < 64 * 192; i += THREADS) {
    int f = i / 192, k = i % 192;
    int sel = k >> 6, d = k & 63;
    const float* Wsrc = (sel == 0) ? Wp : (sel == 1 ? Wl : Wr);
    sW[f * WPAD + k] = (f16)Wsrc[f * 64 + d];
  }
  for (int i = tid; i < 64 * 64; i += THREADS) {
    int o = i >> 6, f = i & 63;
    sP[o * PPAD + f] = (f16)Wproj[i];
  }
  if (tid < 64) { sBias[tid] = bias[tid]; sBproj[tid] = bproj[tid]; }
  __syncthreads();

  const int  lane = tid & 31;
  const int  wave = tid >> 5;
  const int  m    = lane & 15;   // A row / B,C column index for this lane
  const int  hi   = lane >> 4;   // half-wave select
  const int  cb0  = hi * 8;      // A-frag K sub-block offset
  const int  koff = hi * 8;      // B-frag K sub-block offset
  const long totalWaves = (long)gridDim.x * WAVES_PER_BLOCK;
  f16* hbuf = sH[wave];

  for (long pair = (long)blockIdx.x * WAVES_PER_BLOCK + wave; pair < numPairs;
       pair += totalWaves) {
    const long nodeBase = pair * 32;
    const long row0 = nodeBase + m;        // tile 0 row for this lane
    const long row1 = nodeBase + 16 + m;   // tile 1 row for this lane
    const long long li0 = lidx[row0], li1 = lidx[row1];
    const long long ri0 = ridx[row0], ri1 = ridx[row1];
    const float* rowX0 = nodes + row0 * 64;
    const float* rowX1 = nodes + row1 * 64;
    const float* rowL0 = nodes + (long)li0 * 64;
    const float* rowL1 = nodes + (long)li1 * 64;
    const float* rowR0 = nodes + (long)ri0 * 64;
    const float* rowR1 = nodes + (long)ri1 * 64;

    // Prefetch next pair's streaming data (wave-uniform branch).
    long np = pair + totalWaves;
    if (np < numPairs) {
      __builtin_prefetch(nodes + (np * 32 + m) * 64 + hi * 32, 0, 3);
      __builtin_prefetch(nodes + (np * 32 + 16 + m) * 64 + hi * 32, 0, 3);
      __builtin_prefetch(lidx + np * 32 + m, 0, 3);
      __builtin_prefetch(ridx + np * 32 + m, 0, 3);
    }

    // ---- build A: 2 x (16 nodes x 192 concat features), 6 f16 frags each ----
    v16h a0[6], a1[6];
    a0[0] = build_a_frag(rowX0, cb0);       a1[0] = build_a_frag(rowX1, cb0);
    a0[1] = build_a_frag(rowX0, 32 + cb0);  a1[1] = build_a_frag(rowX1, 32 + cb0);
    a0[2] = build_a_frag(rowL0, cb0);       a1[2] = build_a_frag(rowL1, cb0);
    a0[3] = build_a_frag(rowL0, 32 + cb0);  a1[3] = build_a_frag(rowL1, 32 + cb0);
    a0[4] = build_a_frag(rowR0, cb0);       a1[4] = build_a_frag(rowR1, cb0);
    a0[5] = build_a_frag(rowR0, 32 + cb0);  a1[5] = build_a_frag(rowR1, 32 + cb0);

    // ---- h = relu(A * Wcat^T + bias): 4 filter tiles x 6 K chunks x 2 M tiles
    #pragma unroll
    for (int t = 0; t < 4; ++t) {
      v8f acc0 = {0.f, 0.f, 0.f, 0.f, 0.f, 0.f, 0.f, 0.f};
      v8f acc1 = {0.f, 0.f, 0.f, 0.f, 0.f, 0.f, 0.f, 0.f};
      const int f = t * 16 + m;
      #pragma unroll
      for (int c = 0; c < 6; ++c) {
        v16h b = load_b_frag(&sW[f * WPAD + c * 32 + koff]);  // shared by both tiles
        acc0 = __builtin_amdgcn_wmma_f32_16x16x32_f16(
            false, a0[c], false, b, (short)0, acc0, false, false);
        acc1 = __builtin_amdgcn_wmma_f32_16x16x32_f16(
            false, a1[c], false, b, (short)0, acc1, false, false);
      }
      const float bv = sBias[f];
      #pragma unroll
      for (int v = 0; v < 8; ++v) {
        float h0 = fmaxf(acc0[v] + bv, 0.0f);
        float h1 = fmaxf(acc1[v] + bv, 0.0f);
        hbuf[(v + hi * 8) * HPAD + f]        = (f16)h0;  // tile0 -> rows 0..15
        hbuf[(16 + v + hi * 8) * HPAD + f]   = (f16)h1;  // tile1 -> rows 16..31
      }
    }

    // Cross-lane LDS handoff within the wave: stop compiler reordering and
    // wait for all DS stores before re-reading in A layout.
    asm volatile("s_wait_dscnt 0" ::: "memory");

    // ---- reload h in A-fragment layout (2 K chunks of 32, per tile) ----
    v16h ha0[2], ha1[2];
    #pragma unroll
    for (int c = 0; c < 2; ++c) {
      const f16* p0 = &hbuf[m * HPAD + c * 32 + cb0];
      const f16* p1 = &hbuf[(16 + m) * HPAD + c * 32 + cb0];
      const v4h* q0 = (const v4h*)p0;
      const v4h* q1 = (const v4h*)(p0 + 16);
      const v4h* r0 = (const v4h*)p1;
      const v4h* r1 = (const v4h*)(p1 + 16);
      ha0[c] = cat4(q0[0], q0[1], q1[0], q1[1]);
      ha1[c] = cat4(r0[0], r0[1], r1[0], r1[1]);
    }

    // ---- out = h * Wproj^T + bproj ----
    #pragma unroll
    for (int t = 0; t < 4; ++t) {
      v8f acc0 = {0.f, 0.f, 0.f, 0.f, 0.f, 0.f, 0.f, 0.f};
      v8f acc1 = {0.f, 0.f, 0.f, 0.f, 0.f, 0.f, 0.f, 0.f};
      const int o = t * 16 + m;
      #pragma unroll
      for (int c = 0; c < 2; ++c) {
        v16h b = load_b_frag(&sP[o * PPAD + c * 32 + koff]);  // shared by both tiles
        acc0 = __builtin_amdgcn_wmma_f32_16x16x32_f16(
            false, ha0[c], false, b, (short)0, acc0, false, false);
        acc1 = __builtin_amdgcn_wmma_f32_16x16x32_f16(
            false, ha1[c], false, b, (short)0, acc1, false, false);
      }
      const float bv = sBproj[o];
      float* op0 = out + nodeBase * 64 + o;
      float* op1 = out + (nodeBase + 16) * 64 + o;
      #pragma unroll
      for (int v = 0; v < 8; ++v) {
        op0[(long)(v + hi * 8) * 64] = acc0[v] + bv;
        op1[(long)(v + hi * 8) * 64] = acc1[v] + bv;
      }
    }
  }
}

// Scalar tail for N % 32 != 0 (N = 1M is divisible; kept for determinism/safety).
__global__ void treeconv_tail(const float* __restrict__ nodes,
                              const long long* __restrict__ lidx,
                              const long long* __restrict__ ridx,
                              const float* __restrict__ Wp,
                              const float* __restrict__ Wl,
                              const float* __restrict__ Wr,
                              const float* __restrict__ bias,
                              const float* __restrict__ Wproj,
                              const float* __restrict__ bproj,
                              float* __restrict__ out, int N, int start)
{
  int n = start + blockIdx.x * blockDim.x + threadIdx.x;
  if (n >= N) return;
  const float* x = nodes + (long)n * 64;
  const float* l = nodes + (long)lidx[n] * 64;
  const float* r = nodes + (long)ridx[n] * 64;
  float h[64];
  for (int f = 0; f < 64; ++f) {
    float s = bias[f];
    for (int d = 0; d < 64; ++d)
      s += Wp[f * 64 + d] * x[d] + Wl[f * 64 + d] * l[d] + Wr[f * 64 + d] * r[d];
    h[f] = fmaxf(s, 0.f);
  }
  for (int o = 0; o < 64; ++o) {
    float s = bproj[o];
    for (int f = 0; f < 64; ++f) s += Wproj[o * 64 + f] * h[f];
    out[(long)n * 64 + o] = s;
  }
}

extern "C" void kernel_launch(void* const* d_in, const int* in_sizes, int n_in,
                              void* d_out, int out_size, void* d_ws, size_t ws_size,
                              hipStream_t stream) {
  const float*     nodes = (const float*)d_in[0];
  const long long* lidx  = (const long long*)d_in[1];
  const long long* ridx  = (const long long*)d_in[2];
  const float*     Wp    = (const float*)d_in[3];
  const float*     Wl    = (const float*)d_in[4];
  const float*     Wr    = (const float*)d_in[5];
  const float*     bias  = (const float*)d_in[6];
  const float*     Wproj = (const float*)d_in[7];
  const float*     bproj = (const float*)d_in[8];
  float*           out   = (float*)d_out;

  const int N = in_sizes[1];        // left_idx has N elements
  const int numPairs = N / 32;

  if (numPairs > 0) {
    int blocks = (numPairs + WAVES_PER_BLOCK - 1) / WAVES_PER_BLOCK;
    if (blocks > 4096) blocks = 4096;
    treeconv_wmma<<<blocks, THREADS, 0, stream>>>(
        nodes, lidx, ridx, Wp, Wl, Wr, bias, Wproj, bproj, out, numPairs);
  }
  const int tailStart = numPairs * 32;
  const int tail = N - tailStart;
  if (tail > 0) {
    treeconv_tail<<<(tail + 63) / 64, 64, 0, stream>>>(
        nodes, lidx, ridx, Wp, Wl, Wr, bias, Wproj, bproj, out, N, tailStart);
  }
}
